// SingleAreaRNN_18382460027621
// MI455X (gfx1250) — matrix operand order repaired and checked by reading, not compiled, not run on hardware
//
#include <hip/hip_runtime.h>

typedef __attribute__((ext_vector_type(16))) __bf16 v16bf;
typedef __attribute__((ext_vector_type(8)))  float  v8f;
typedef unsigned short ushort_t;

#define ALPHA_C 0.1f
#define NOISE_C 0.02f
#define B_SZ 64
#define T_SZ 512
#define I_SZ 128
#define H_SZ 1024
#define O_SZ 64
#define NWG 16
#define HS (H_SZ / NWG)   // 64 hidden columns per workgroup
#define NTHREADS 256      // 8 waves (wave32)

union Frag { v16bf v; uint4 u[2]; };

__device__ __forceinline__ ushort_t f2bf(float f) {
    unsigned u = __float_as_uint(f);
    unsigned r = (u + 0x7FFFu + ((u >> 16) & 1u)) >> 16;   // round-to-nearest-even
    return (ushort_t)r;
}

__device__ __forceinline__ float fast_tanh(float x) {
#if __has_builtin(__builtin_amdgcn_tanhf)
    return __builtin_amdgcn_tanhf(x);      // v_tanh_f32 (gfx1250 TRANS op)
#else
    return tanhf(x);
#endif
}

__device__ __forceinline__ void wait_asynccnt0() {
#if __has_builtin(__builtin_amdgcn_s_wait_asynccnt)
    __builtin_amdgcn_s_wait_asynccnt(0);
#else
    asm volatile("s_wait_asynccnt 0x0" ::: "memory");
#endif
}

// ASYNCcnt-tracked copy: 16 bytes per lane, global -> LDS (CDNA5 async path)
__device__ __forceinline__ void async_copy_b128(unsigned lds_byte_addr, const void* gsrc) {
    asm volatile("global_load_async_to_lds_b128 %0, %1, off"
                 :: "v"(lds_byte_addr), "v"((unsigned long long)(size_t)gsrc)
                 : "memory");
}

// B operand 32x16 bf16: lane = column (lane&15), K halves at hi*8 and 16+hi*8
__device__ __forceinline__ Frag load_b_frag(const ushort_t* base, int ldb,
                                            int n0, int k0, int laneM, int hi) {
    Frag f;
    const ushort_t* p = base + (size_t)(n0 + laneM) * ldb + k0 + hi * 8;
    f.u[0] = *(const uint4*)(p);
    f.u[1] = *(const uint4*)(p + 16);
    return f;
}

// A operand 16x32 bf16: lanes 0-15 rows M, K halves at hi*8 and 16+hi*8
__device__ __forceinline__ Frag load_a_frag(const ushort_t* base, int ld,
                                            int m0, int k0, int laneM, int hi) {
    Frag f;
    const ushort_t* p = base + (size_t)(m0 + laneM) * ld + k0 + hi * 8;
    f.u[0] = *(const uint4*)(p);
    f.u[1] = *(const uint4*)(p + 16);
    return f;
}

#define WMMA_BF16(A, Bv, C) \
    __builtin_amdgcn_wmma_f32_16x16x32_bf16(false, (A), false, (Bv), (short)0, (C), false, false)

// ---------------------------------------------------------------------------
// Prep kernel 1: zero d_out, tanh(h) double-buffer, barrier counter
// ---------------------------------------------------------------------------
__global__ void zero_init_kernel(float* __restrict__ out, ushort_t* __restrict__ thbuf,
                                 unsigned* __restrict__ bar) {
    size_t i = (size_t)blockIdx.x * blockDim.x + threadIdx.x;
    size_t stride = (size_t)gridDim.x * blockDim.x;
    const size_t nout = (size_t)B_SZ * T_SZ * O_SZ;
    const size_t nth  = (size_t)2 * B_SZ * H_SZ;
    for (size_t j = i; j < nout; j += stride) out[j] = 0.0f;
    for (size_t j = i; j < nth;  j += stride) thbuf[j] = 0;
    if (i == 0) *bar = 0u;
}

// ---------------------------------------------------------------------------
// Prep kernel 2: fp32 -> bf16 conversions (one-time, off the scan's hot path)
//   xbf  [b][t][i] = x[b][t][i]          (B x T x I, same layout)
//   wrecT[c][r]    = wrec[r][c]          (H x H, B-operand column-major)
//   wiT  [h][i]    = wi[i][h]            (H x I)
//   woT  [o][h]    = wo[h][o]            (O x H)
// ---------------------------------------------------------------------------
__global__ void prep_kernel(const float* __restrict__ x, const float* __restrict__ wi,
                            const float* __restrict__ wrec, const float* __restrict__ wo,
                            ushort_t* __restrict__ xbf, ushort_t* __restrict__ wiT,
                            ushort_t* __restrict__ wrecT, ushort_t* __restrict__ woT) {
    size_t idx = (size_t)blockIdx.x * blockDim.x + threadIdx.x;
    size_t stride = (size_t)gridDim.x * blockDim.x;
    const size_t nx = (size_t)B_SZ * T_SZ * I_SZ;
    for (size_t j = idx; j < nx; j += stride) xbf[j] = f2bf(x[j]);
    for (size_t j = idx; j < (size_t)H_SZ * H_SZ; j += stride) {
        size_t c = j / H_SZ, r = j % H_SZ;
        wrecT[c * H_SZ + r] = f2bf(wrec[r * H_SZ + c]);
    }
    for (size_t j = idx; j < (size_t)H_SZ * I_SZ; j += stride) {
        size_t h = j / I_SZ, i2 = j % I_SZ;
        wiT[h * I_SZ + i2] = f2bf(wi[i2 * H_SZ + h]);
    }
    for (size_t j = idx; j < (size_t)O_SZ * H_SZ; j += stride) {
        size_t o = j / H_SZ, h2 = j % H_SZ;
        woT[o * H_SZ + h2] = f2bf(wo[h2 * O_SZ + o]);
    }
}

// ---------------------------------------------------------------------------
// Persistent scan kernel: 16 WGs x 256 threads. WG g owns hidden cols
// [g*64, g*64+64). h stays in VGPRs (WMMA C layout) for all 512 steps.
// Per step: async-stage tanh(h_prev) [64x1024] bf16 into LDS while x@wi
// runs, then K=1024 recurrent WMMA sweep entirely from LDS.
// ---------------------------------------------------------------------------
__global__ __launch_bounds__(NTHREADS) void rnn_scan_kernel(
    const ushort_t* __restrict__ xbf,   // [B][T][I] bf16
    const float* __restrict__ noise,    // [B][T][H] fp32
    const ushort_t* __restrict__ wiT,   // [H][I] bf16
    const ushort_t* __restrict__ wrecT, // [H][H] bf16 (col-major of wrec)
    const ushort_t* __restrict__ woT,   // [O][H] bf16
    ushort_t* __restrict__ thbuf,       // [2][B][H] bf16 double buffer
    unsigned* __restrict__ bar,
    float* __restrict__ out)            // [B][T][O] fp32
{
    extern __shared__ ushort_t smem[];
    ushort_t* lds_wrec  = smem;                         // [HS][H]  128 KB
    ushort_t* lds_wi    = lds_wrec + HS * H_SZ;         // [HS][I]   16 KB
    ushort_t* lds_wo    = lds_wi   + HS * I_SZ;         // [O][HS]    8 KB
    ushort_t* lds_th    = lds_wo   + O_SZ * HS;         // [64][HS]   8 KB
    ushort_t* lds_stage = lds_th   + 64 * HS;           // [B][H]   128 KB

    const int g   = blockIdx.x;          // 0..15
    const int hb  = g * HS;
    const int tid = threadIdx.x;
    const int wave  = tid >> 5;
    const int lane  = tid & 31;
    const int laneM = lane & 15;
    const int hi    = lane >> 4;
    const int tm  = wave >> 1;           // batch row tile (0..3)
    const int tn0 = (wave & 1) * 2;      // column tiles within slice
    const int tn1 = tn0 + 1;
    const int brow = tm * 16 + laneM;    // A-operand batch row for this lane

    // LDS byte address of the staging buffer (low 32 bits of generic LDS ptr)
    const unsigned stage_base = (unsigned)(size_t)(void*)lds_stage;

    // ---- stage this WG's weight slices into LDS (once) ----
    {
        const uint4* src = (const uint4*)(wrecT + (size_t)hb * H_SZ);
        uint4* dst = (uint4*)lds_wrec;
        for (int i = tid; i < HS * H_SZ / 8; i += NTHREADS) dst[i] = src[i];
        const uint4* src2 = (const uint4*)(wiT + (size_t)hb * I_SZ);
        uint4* dst2 = (uint4*)lds_wi;
        for (int i = tid; i < HS * I_SZ / 8; i += NTHREADS) dst2[i] = src2[i];
        for (int i = tid; i < O_SZ * HS / 8; i += NTHREADS) {
            int row = i / (HS / 8), part = i % (HS / 8);
            ((uint4*)lds_wo)[i] = *(const uint4*)(woT + (size_t)row * H_SZ + hb + part * 8);
        }
    }
    __syncthreads();

    v8f h0 = {}; v8f h1 = {};            // persistent fp32 hidden-state tiles

    for (int t = 0; t < T_SZ; ++t) {
        const int wb = t & 1, rb = wb ^ 1;
        const ushort_t* th_r = thbuf + (size_t)rb * (B_SZ * H_SZ);
        ushort_t*       th_w = thbuf + (size_t)wb * (B_SZ * H_SZ);

        // ---- kick off async stage of tanh(h_prev) [64][1024] bf16 -> LDS ----
        // 128 KB / (256 threads * 16 B) = 32 b128 transfers per thread.
        {
            const ushort_t* src = th_r + (size_t)tid * 8;   // 16 B per lane per shot
            unsigned dst = stage_base + (unsigned)tid * 16u;
            #pragma unroll 8
            for (int k = 0; k < (B_SZ * H_SZ) / (NTHREADS * 8); ++k) {
                async_copy_b128(dst, src);
                src += NTHREADS * 8;        // 256 threads * 8 ushorts
                dst += NTHREADS * 16u;
            }
        }

        v8f acc0 = {}; v8f acc1 = {};

        // ---- x_t @ wi (K = 128, bf16 pre-converted) -- overlaps async stage ----
        {
            const ushort_t* xrow = xbf + (size_t)brow * (T_SZ * I_SZ) + (size_t)t * I_SZ;
            #pragma unroll
            for (int kk = 0; kk < I_SZ / 32; ++kk) {
                Frag a;
                const ushort_t* p = xrow + kk * 32 + hi * 8;
                a.u[0] = *(const uint4*)(p);
                a.u[1] = *(const uint4*)(p + 16);
                Frag b0 = load_b_frag(lds_wi, I_SZ, tn0 * 16, kk * 32, laneM, hi);
                Frag b1 = load_b_frag(lds_wi, I_SZ, tn1 * 16, kk * 32, laneM, hi);
                acc0 = WMMA_BF16(a.v, b0.v, acc0);
                acc1 = WMMA_BF16(a.v, b1.v, acc1);
            }
        }

        // ---- wait for staged A data, then K=1024 recurrent sweep from LDS ----
        wait_asynccnt0();
        __syncthreads();
        {
            const ushort_t* arow = lds_stage + (size_t)brow * H_SZ;
            #pragma unroll 4
            for (int kk = 0; kk < H_SZ / 32; ++kk) {
                Frag a;
                const ushort_t* p = arow + kk * 32 + hi * 8;
                a.u[0] = *(const uint4*)(p);
                a.u[1] = *(const uint4*)(p + 16);
                Frag b0 = load_b_frag(lds_wrec, H_SZ, tn0 * 16, kk * 32, laneM, hi);
                Frag b1 = load_b_frag(lds_wrec, H_SZ, tn1 * 16, kk * 32, laneM, hi);
                acc0 = WMMA_BF16(a.v, b0.v, acc0);
                acc1 = WMMA_BF16(a.v, b1.v, acc1);
            }
        }

        // ---- leaky update + noise + tanh; publish tanh(h) (bf16) ----
        {
            const float* nbase = noise + (size_t)t * H_SZ + hb;
            const int c0 = tn0 * 16 + laneM, c1 = tn1 * 16 + laneM;
            #pragma unroll
            for (int r = 0; r < 8; ++r) {
                int b_ = tm * 16 + r + 8 * hi;          // C-layout row
                float n0 = nbase[(size_t)b_ * (T_SZ * H_SZ) + c0];
                float n1 = nbase[(size_t)b_ * (T_SZ * H_SZ) + c1];
                h0[r] = (1.0f - ALPHA_C) * h0[r] + ALPHA_C * (acc0[r] + NOISE_C * n0);
                h1[r] = (1.0f - ALPHA_C) * h1[r] + ALPHA_C * (acc1[r] + NOISE_C * n1);
                float t0 = fast_tanh(h0[r]);
                float t1 = fast_tanh(h1[r]);
                ushort_t bv0 = f2bf(t0), bv1 = f2bf(t1);
                th_w[(size_t)b_ * H_SZ + hb + c0] = bv0;
                th_w[(size_t)b_ * H_SZ + hb + c1] = bv1;
                lds_th[b_ * HS + c0] = bv0;
                lds_th[b_ * HS + c1] = bv1;
            }
        }
        __syncthreads();   // lds_th complete for all waves

        // ---- partial y_t = tanh(h)_slice @ wo_slice, reduce across WGs ----
        {
            v8f oa0 = {}; v8f oa1 = {};
            #pragma unroll
            for (int kk = 0; kk < HS / 32; ++kk) {
                Frag a  = load_a_frag(lds_th, HS, tm * 16, kk * 32, laneM, hi);
                Frag b0 = load_b_frag(lds_wo, HS, tn0 * 16, kk * 32, laneM, hi);
                Frag b1 = load_b_frag(lds_wo, HS, tn1 * 16, kk * 32, laneM, hi);
                oa0 = WMMA_BF16(a.v, b0.v, oa0);
                oa1 = WMMA_BF16(a.v, b1.v, oa1);
            }
            float* orow = out + (size_t)t * O_SZ;
            const int o0 = tn0 * 16 + laneM, o1 = tn1 * 16 + laneM;
            #pragma unroll
            for (int r = 0; r < 8; ++r) {
                int b_ = tm * 16 + r + 8 * hi;
                atomicAdd(orow + (size_t)b_ * (T_SZ * O_SZ) + o0, oa0[r]);
                atomicAdd(orow + (size_t)b_ * (T_SZ * O_SZ) + o1, oa1[r]);
            }
        }

        // ---- device-wide step barrier (monotonic counter, no sense flip) ----
        __threadfence();
        __syncthreads();
        if (tid == 0) {
            __hip_atomic_fetch_add(bar, 1u, __ATOMIC_ACQ_REL, __HIP_MEMORY_SCOPE_AGENT);
            unsigned target = (unsigned)(t + 1) * NWG;
            while (__hip_atomic_load(bar, __ATOMIC_ACQUIRE, __HIP_MEMORY_SCOPE_AGENT) < target) {
                __builtin_amdgcn_s_sleep(1);
            }
        }
        __syncthreads();
    }
}

// ---------------------------------------------------------------------------
extern "C" void kernel_launch(void* const* d_in, const int* in_sizes, int n_in,
                              void* d_out, int out_size, void* d_ws, size_t ws_size,
                              hipStream_t stream) {
    const float* x     = (const float*)d_in[0];   // [B,T,I]
    const float* noise = (const float*)d_in[1];   // [B,T,H]
    const float* wi    = (const float*)d_in[2];   // [I,H]
    const float* wrec  = (const float*)d_in[3];   // [H,H]
    const float* wo    = (const float*)d_in[4];   // [H,O]
    float* out = (float*)d_out;                   // [B,T,O]

    // workspace layout (bf16 stored as ushort): ~11.2 MB total
    ushort_t* wrecT = (ushort_t*)d_ws;                     // H*H
    ushort_t* wiT   = wrecT + (size_t)H_SZ * H_SZ;         // H*I
    ushort_t* woT   = wiT   + (size_t)H_SZ * I_SZ;         // O*H
    ushort_t* thbuf = woT   + (size_t)O_SZ * H_SZ;         // 2*B*H
    ushort_t* xbf   = thbuf + (size_t)2 * B_SZ * H_SZ;     // B*T*I
    unsigned* bar   = (unsigned*)(xbf + (size_t)B_SZ * T_SZ * I_SZ);

    zero_init_kernel<<<1024, 256, 0, stream>>>(out, thbuf, bar);
    prep_kernel<<<4096, 256, 0, stream>>>(x, wi, wrec, wo, xbf, wiT, wrecT, woT);

    const size_t smem_bytes =
        (size_t)(HS * H_SZ + HS * I_SZ + O_SZ * HS + 64 * HS + B_SZ * H_SZ)
        * sizeof(ushort_t); // 288 KB
    rnn_scan_kernel<<<NWG, NTHREADS, smem_bytes, stream>>>(
        xbf, noise, wiT, wrecT, woT, thbuf, bar, out);
}
